// MTLPLEMOEModel_37503654429104
// MI455X (gfx1250) — compile-verified, matrix operand order checked by reading, not run on hardware
//
#include <hip/hip_runtime.h>

// ---------------- problem dims ----------------
constexpr int B_ = 4096, N_ = 64, K_ = 48, D_ = 64;
constexpr int F_ = N_ * D_;            // 4096
constexpr int E_ = 8, H_ = 1024, O_ = 512;
constexpr int T_ = 2, TH_ = 256;

typedef __attribute__((ext_vector_type(16))) _Float16 v16h;
typedef __attribute__((ext_vector_type(8)))  _Float16 v8h;
typedef __attribute__((ext_vector_type(8)))  float    v8f;
typedef __attribute__((ext_vector_type(4)))  int      v4i;

// ---------------- GEMM tiling (wave32, 8 waves / block) ----------------
constexpr int BM = 64, BN = 128, BK = 32;
constexpr int PITCH = 40;   // LDS row pitch in halves: 80B rows -> 16B-aligned b128, no bank cycling

// CDNA5 async Global->LDS path (ASYNCcnt), guarded so the file compiles on any toolchain.
#if defined(__has_builtin)
#if __has_builtin(__builtin_amdgcn_global_load_async_to_lds_b128) && \
    __has_builtin(__builtin_amdgcn_s_wait_asynccnt)
#define USE_ASYNC_LDS 1
#endif
#endif
#ifndef USE_ASYNC_LDS
#define USE_ASYNC_LDS 0
#endif

#define GAS __attribute__((address_space(1)))
#define LAS __attribute__((address_space(3)))

// ---------------- workspace layout (bytes) ----------------
constexpr size_t OFF_FLAT16 = 0;                                   // B*F f16
constexpr size_t OFF_W1T    = OFF_FLAT16 + (size_t)B_ * F_ * 2;    // E*H*F f16 (transposed)
constexpr size_t OFF_W2T    = OFF_W1T + (size_t)E_ * H_ * F_ * 2;  // E*O*H f16
constexpr size_t OFF_TW1T   = OFF_W2T + (size_t)E_ * O_ * H_ * 2;  // T*TH*O f16
constexpr size_t OFF_H16    = OFF_TW1T + (size_t)T_ * TH_ * O_ * 2;// 2B slots * H f16
constexpr size_t OFF_MOE    = OFF_H16 + (size_t)2 * B_ * H_ * 2;   // B*O f32
constexpr size_t OFF_MOE16  = OFF_MOE + (size_t)B_ * O_ * 4;       // B*O f16
constexpr size_t OFF_TH16   = OFF_MOE16 + (size_t)B_ * O_ * 2;     // B*T*TH f16
constexpr size_t OFF_ROUTE  = OFF_TH16 + (size_t)B_ * T_ * TH_ * 2;// B * float4
constexpr size_t OFF_ROWS   = OFF_ROUTE + (size_t)B_ * 16;         // 2B ints
constexpr size_t OFF_GW     = OFF_ROWS + (size_t)2 * B_ * 4;       // 2B floats
constexpr size_t OFF_CNT    = OFF_GW + (size_t)2 * B_ * 4;         // E ints
constexpr size_t OFF_OFFS   = OFF_CNT + 64;                        // E+1 ints
constexpr size_t OFF_CUR    = OFF_OFFS + 64;                       // E ints

// ---------------- WMMA helpers ----------------
__device__ __forceinline__ v8f wmma_f16(v16h a, v16h b, v8f c) {
  // emits v_wmma_f32_16x16x32_f16
  return __builtin_amdgcn_wmma_f32_16x16x32_f16(false, a, false, b, (short)0, c, false, false);
}

// Load a 16x32 f16 fragment from LDS (row pitch = PITCH halves) per ISA 7.12.2:
// lanes 0-15 hold row (lane&15), K = {0..7, 16..23}; lanes 16-31: K = {8..15, 24..31}.
__device__ __forceinline__ v16h ldsFrag(const _Float16* base, int r0) {
  int lane = threadIdx.x & 31;
  const _Float16* p = base + (r0 + (lane & 15)) * PITCH + ((lane >> 4) << 3);
  v8h lo = *(const v8h*)p;
  v8h hi = *(const v8h*)(p + 16);
  v16h f;
#pragma unroll
  for (int i = 0; i < 8; ++i) { f[i] = lo[i]; f[8 + i] = hi[i]; }
  return f;
}

// ---------------- shared 64x128x(KDIM) WMMA GEMM core ----------------
// aSrc: this thread's staging pointer into A (row ar = t>>2, 8 halves at aseg)
// bSrc: this thread's staging pointer into B^T (row bn = t>>1, 16 halves at bseg)
// Accumulates the calling wave's 4 C tiles (16x16 each) over KDIM.
template <int KDIM>
__device__ __forceinline__ void gemm_core(const _Float16* __restrict__ aSrc,
                                          const _Float16* __restrict__ bSrc,
                                          v8f acc[4]) {
  int t = threadIdx.x;
  int ar = t >> 2, aseg = (t & 3) * 8;
  int bn = t >> 1, bseg = (t & 1) * 16;
  int w = t >> 5;
  int mw = (w & 3) * 16, nw = (w >> 2) * 64;

#if USE_ASYNC_LDS
  // Double-buffered: DMA writes LDS directly (no VGPR round-trip); loads for
  // step k+1 overlap the WMMAs of step k. One s_wait_asynccnt + one barrier/step.
  __shared__ _Float16 As[2][BM * PITCH];
  __shared__ _Float16 Bs[2][BN * PITCH];
  auto stage = [&](int buf, int k0) {
    __builtin_amdgcn_global_load_async_to_lds_b128(
        (GAS v4i*)(aSrc + k0), (LAS v4i*)&As[buf][ar * PITCH + aseg], 0, 0);
    __builtin_amdgcn_global_load_async_to_lds_b128(
        (GAS v4i*)(bSrc + k0), (LAS v4i*)&Bs[buf][bn * PITCH + bseg], 0, 0);
    __builtin_amdgcn_global_load_async_to_lds_b128(
        (GAS v4i*)(bSrc + k0 + 8), (LAS v4i*)&Bs[buf][bn * PITCH + bseg + 8], 0, 0);
  };
  stage(0, 0);
  int cur = 0;
  for (int k0 = 0; k0 < KDIM; k0 += BK) {
    __builtin_amdgcn_s_wait_asynccnt(0);
    __syncthreads();
    if (k0 + BK < KDIM) stage(cur ^ 1, k0 + BK);
    v16h af = ldsFrag(As[cur], mw);
#pragma unroll
    for (int j = 0; j < 4; ++j) {
      v16h bf = ldsFrag(Bs[cur], nw + j * 16);
      acc[j] = wmma_f16(af, bf, acc[j]);
    }
    cur ^= 1;
  }
#else
  // Sync fallback: single buffer, global->VGPR->LDS staging, prefetch next step.
  __shared__ _Float16 As[BM * PITCH];
  __shared__ _Float16 Bs[BN * PITCH];
  for (int k0 = 0; k0 < KDIM; k0 += BK) {
    __syncthreads();
    *(v8h*)&As[ar * PITCH + aseg] = *(const v8h*)(aSrc + k0);
    *(v8h*)&Bs[bn * PITCH + bseg] = *(const v8h*)(bSrc + k0);
    *(v8h*)&Bs[bn * PITCH + bseg + 8] = *(const v8h*)(bSrc + k0 + 8);
    __builtin_prefetch(aSrc + k0 + BK, 0, 1);   // global_prefetch_b8 next K-step
    __builtin_prefetch(bSrc + k0 + BK, 0, 1);
    __syncthreads();
    v16h af = ldsFrag(As, mw);
#pragma unroll
    for (int j = 0; j < 4; ++j) {
      v16h bf = ldsFrag(Bs, nw + j * 16);
      acc[j] = wmma_f16(af, bf, acc[j]);
    }
  }
#endif
}

// ---------------- init ----------------
__global__ void init_kernel(float* __restrict__ moe, int* __restrict__ counts) {
  int idx = blockIdx.x * 256 + threadIdx.x;
  if (idx < B_ * O_) moe[idx] = 0.f;
  if (idx < E_) counts[idx] = 0;
}

// ---------------- fp32 -> f16 transposed weight conversion ----------------
// src: mats x R x C (row-major) ; dst: mats x C x R
__global__ void transpose_cvt_kernel(const float* __restrict__ src, _Float16* __restrict__ dst,
                                     int R, int C) {
  __shared__ _Float16 tile[64 * 66];
  int mat = blockIdx.z;
  int c0 = blockIdx.x * 64;
  int r0 = blockIdx.y * 64;
  const float* s = src + (size_t)mat * R * C;
  _Float16* d = dst + (size_t)mat * R * C;
  int t = threadIdx.x;
  int cc = t & 63;
  for (int rr = t >> 6; rr < 64; rr += 4)
    tile[cc * 66 + rr] = (_Float16)s[(size_t)(r0 + rr) * C + c0 + cc];
  __syncthreads();
  int cw = t >> 2;
  int seg = (t & 3) * 16;
  _Float16* out = d + (size_t)(c0 + cw) * R + r0 + seg;
#pragma unroll
  for (int i = 0; i < 16; ++i) out[i] = tile[cw * 66 + seg + i];
}

// ---------------- PLE + per-feature embedding (fused) ----------------
__global__ void ple_embed_kernel(const float* __restrict__ x, const float* __restrict__ plw,
                                 const float* __restrict__ plb, const float* __restrict__ embW,
                                 const float* __restrict__ embB, _Float16* __restrict__ flat16) {
  __shared__ float enc[N_ * K_];
  int b = blockIdx.x;
  int t = threadIdx.x;
  for (int idx = t; idx < N_ * K_; idx += 256) {
    int n = idx / K_, k = idx - n * K_;
    float xv = x[b * N_ + n];
    float v = plb[idx] + plw[idx] * xv;
    if (k == 0)           v = fminf(v, 1.f);
    else if (k == K_ - 1) v = fmaxf(v, 0.f);
    else                  v = fminf(fmaxf(v, 0.f), 1.f);
    enc[idx] = v;
  }
  __syncthreads();
  for (int idx = t; idx < F_; idx += 256) {
    int n = idx >> 6, d = idx & 63;
    float acc = embB[n * D_ + d];
    const float* wp = embW + (size_t)n * K_ * D_ + d;
    const float* ep = enc + n * K_;
#pragma unroll 4
    for (int k = 0; k < K_; ++k) acc += ep[k] * wp[(size_t)k * D_];
    flat16[(size_t)b * F_ + idx] = (_Float16)fmaxf(acc, 0.f);
  }
}

// ---------------- gate: logits, softmax, top-2, renorm + expert counts ----------------
__global__ void gate_kernel(const _Float16* __restrict__ flat16, const float* __restrict__ gW,
                            const float* __restrict__ gB, float4* __restrict__ route,
                            int* __restrict__ counts) {
  int w = threadIdx.x >> 5, lane = threadIdx.x & 31;
  int b = blockIdx.x * 8 + w;
  const _Float16* fr = flat16 + (size_t)b * F_;
  float acc[E_];
#pragma unroll
  for (int e = 0; e < E_; ++e) acc[e] = 0.f;
  for (int it = 0; it < F_ / 256; ++it) {
    int f0 = it * 256 + lane * 8;
    v8h xv = *(const v8h*)(fr + f0);
#pragma unroll
    for (int i = 0; i < 8; ++i) {
      float xf = (float)xv[i];
      const float* wr = gW + (size_t)(f0 + i) * E_;
#pragma unroll
      for (int e = 0; e < E_; ++e) acc[e] += xf * wr[e];
    }
  }
#pragma unroll
  for (int e = 0; e < E_; ++e)
    for (int m = 16; m >= 1; m >>= 1) acc[e] += __shfl_xor(acc[e], m, 32);
  if (lane == 0) {
    float p[E_], mx = -1e30f;
#pragma unroll
    for (int e = 0; e < E_; ++e) { p[e] = acc[e] + gB[e]; mx = fmaxf(mx, p[e]); }
#pragma unroll
    for (int e = 0; e < E_; ++e) p[e] = __expf(p[e] - mx);
    int i1 = 0;
#pragma unroll
    for (int e = 1; e < E_; ++e) if (p[e] > p[i1]) i1 = e;
    int i2 = (i1 == 0) ? 1 : 0;
#pragma unroll
    for (int e = 0; e < E_; ++e) if (e != i1 && p[e] > p[i2]) i2 = e;
    float g1 = p[i1], g2 = p[i2], inv = 1.f / (g1 + g2);
    g1 *= inv; g2 *= inv;
    atomicAdd(&counts[i1], 1);
    atomicAdd(&counts[i2], 1);
    route[b] = make_float4(__int_as_float(i1), __int_as_float(i2), g1, g2);
  }
}

// ---------------- tiny prefix over E=8 counts ----------------
__global__ void offsets_kernel(const int* __restrict__ counts, int* __restrict__ offs,
                               int* __restrict__ cursor) {
  if (threadIdx.x == 0) {
    int run = 0;
    for (int e = 0; e < E_; ++e) { offs[e] = run; cursor[e] = run; run += counts[e]; }
    offs[E_] = run;
  }
}

// ---------------- scatter rows into per-expert compacted lists ----------------
__global__ void scatter_kernel(const float4* __restrict__ route, int* __restrict__ cursor,
                               int* __restrict__ rowsArr, float* __restrict__ gwArr) {
  int b = blockIdx.x * 256 + threadIdx.x;
  if (b >= B_) return;
  float4 r = route[b];
  int e1 = __float_as_int(r.x), e2 = __float_as_int(r.y);
  int p1 = atomicAdd(&cursor[e1], 1); rowsArr[p1] = b; gwArr[p1] = r.z;
  int p2 = atomicAdd(&cursor[e2], 1); rowsArr[p2] = b; gwArr[p2] = r.w;
}

// ---------------- FC1 routed GEMM: h = relu(flat[rows] @ W1[e] + b1[e]) ----------------
__global__ void fc1_kernel(const _Float16* __restrict__ A, const _Float16* __restrict__ W1T,
                           const float* __restrict__ b1, const int* __restrict__ counts,
                           const int* __restrict__ offs, const int* __restrict__ rowsArr,
                           _Float16* __restrict__ h16) {
  int e = blockIdx.z;
  int cnt = counts[e];
  int m0 = blockIdx.y * BM;
  if (m0 >= cnt) return;
  int off = offs[e];
  int n0 = blockIdx.x * BN;
  int t = threadIdx.x;
  int ar = t >> 2, aseg = (t & 3) * 8;
  int lr = m0 + ar; if (lr >= cnt) lr = cnt - 1;
  const _Float16* aSrc = A + (size_t)rowsArr[off + lr] * F_ + aseg;   // gathered row
  int bn = t >> 1, bseg = (t & 1) * 16;
  const _Float16* bSrc = W1T + ((size_t)e * H_ + n0 + bn) * F_ + bseg;

  v8f z = {0.f, 0.f, 0.f, 0.f, 0.f, 0.f, 0.f, 0.f};
  v8f acc[4];
#pragma unroll
  for (int j = 0; j < 4; ++j) acc[j] = z;
  gemm_core<F_>(aSrc, bSrc, acc);

  int w = t >> 5;
  int mw = (w & 3) * 16, nw = (w >> 2) * 64;
  int lane = t & 31;
  int rowoff = (lane >> 4) * 8, colq = lane & 15;
  const float* b1p = b1 + (size_t)e * H_;
#pragma unroll
  for (int j = 0; j < 4; ++j) {
    int col = n0 + nw + j * 16 + colq;
    float bias = b1p[col];
#pragma unroll
    for (int i = 0; i < 8; ++i) {
      int r = m0 + mw + rowoff + i;
      if (r < cnt) {
        float v = acc[j][i] + bias;
        h16[(size_t)(off + r) * H_ + col] = (_Float16)(v > 0.f ? v : 0.f);
      }
    }
  }
}

// ---------------- FC2 routed GEMM + gated scatter-add into moe ----------------
__global__ void fc2_kernel(const _Float16* __restrict__ h16, const _Float16* __restrict__ W2T,
                           const float* __restrict__ b2, const int* __restrict__ counts,
                           const int* __restrict__ offs, const int* __restrict__ rowsArr,
                           const float* __restrict__ gwArr, float* __restrict__ moe) {
  int e = blockIdx.z;
  int cnt = counts[e];
  int m0 = blockIdx.y * BM;
  if (m0 >= cnt) return;
  int off = offs[e];
  int n0 = blockIdx.x * BN;
  int t = threadIdx.x;
  int ar = t >> 2, aseg = (t & 3) * 8;
  int lr = m0 + ar; if (lr >= cnt) lr = cnt - 1;
  const _Float16* aSrc = h16 + (size_t)(off + lr) * H_ + aseg;
  int bn = t >> 1, bseg = (t & 1) * 16;
  const _Float16* bSrc = W2T + ((size_t)e * O_ + n0 + bn) * H_ + bseg;

  v8f z = {0.f, 0.f, 0.f, 0.f, 0.f, 0.f, 0.f, 0.f};
  v8f acc[4];
#pragma unroll
  for (int j = 0; j < 4; ++j) acc[j] = z;
  gemm_core<H_>(aSrc, bSrc, acc);

  int w = t >> 5;
  int mw = (w & 3) * 16, nw = (w >> 2) * 64;
  int lane = t & 31;
  int rowoff = (lane >> 4) * 8, colq = lane & 15;
  const float* b2p = b2 + (size_t)e * O_;
#pragma unroll
  for (int i = 0; i < 8; ++i) {
    int r = m0 + mw + rowoff + i;
    if (r >= cnt) continue;
    int slot = off + r;
    int brow = rowsArr[slot];
    float g = gwArr[slot];
#pragma unroll
    for (int j = 0; j < 4; ++j) {
      int col = n0 + nw + j * 16 + colq;
      float v = acc[j][i] + b2p[col];
      atomicAdd(&moe[(size_t)brow * O_ + col], g * v);   // exactly 2 adds/elem: bit-deterministic
    }
  }
}

// ---------------- moe fp32 -> f16 ----------------
__global__ void cvt_moe_kernel(const float* __restrict__ moe, _Float16* __restrict__ moe16) {
  int idx = blockIdx.x * 256 + threadIdx.x;
  if (idx < B_ * O_) moe16[idx] = (_Float16)moe[idx];
}

// ---------------- tower layer-1 GEMM: th = relu(moe @ tw1[t] + tb1[t]) ----------------
__global__ void tower_kernel(const _Float16* __restrict__ moe16, const _Float16* __restrict__ tw1T,
                             const float* __restrict__ tb1, _Float16* __restrict__ th16) {
  int task = blockIdx.z;
  int m0 = blockIdx.y * BM;
  int n0 = blockIdx.x * BN;
  int t = threadIdx.x;
  int ar = t >> 2, aseg = (t & 3) * 8;
  const _Float16* aSrc = moe16 + (size_t)(m0 + ar) * O_ + aseg;
  int bn = t >> 1, bseg = (t & 1) * 16;
  const _Float16* bSrc = tw1T + ((size_t)task * TH_ + n0 + bn) * O_ + bseg;

  v8f z = {0.f, 0.f, 0.f, 0.f, 0.f, 0.f, 0.f, 0.f};
  v8f acc[4];
#pragma unroll
  for (int j = 0; j < 4; ++j) acc[j] = z;
  gemm_core<O_>(aSrc, bSrc, acc);

  int w = t >> 5;
  int mw = (w & 3) * 16, nw = (w >> 2) * 64;
  int lane = t & 31;
  int rowoff = (lane >> 4) * 8, colq = lane & 15;
  const float* tb1p = tb1 + (size_t)task * TH_;
#pragma unroll
  for (int j = 0; j < 4; ++j) {
    int col = n0 + nw + j * 16 + colq;
    float bias = tb1p[col];
#pragma unroll
    for (int i = 0; i < 8; ++i) {
      int r = m0 + mw + rowoff + i;
      float v = acc[j][i] + bias;
      th16[((size_t)r * T_ + task) * TH_ + col] = (_Float16)(v > 0.f ? v : 0.f);
    }
  }
}

// ---------------- tower layer-2: 256-dot per (b, task) ----------------
__global__ void final_kernel(const _Float16* __restrict__ th16, const float* __restrict__ tw2,
                             const float* __restrict__ tb2, float* __restrict__ out) {
  int w = threadIdx.x >> 5, lane = threadIdx.x & 31;
  int gid = blockIdx.x * 8 + w;          // == b*T + task
  int task = gid & 1;
  const _Float16* hp = th16 + (size_t)gid * TH_;
  const float* wp = tw2 + task * TH_;
  float acc = 0.f;
#pragma unroll
  for (int s = lane; s < TH_; s += 32) acc += (float)hp[s] * wp[s];
  for (int m = 16; m >= 1; m >>= 1) acc += __shfl_xor(acc, m, 32);
  if (lane == 0) out[gid] = acc + tb2[task];
}

// ---------------- launch ----------------
extern "C" void kernel_launch(void* const* d_in, const int* in_sizes, int n_in,
                              void* d_out, int out_size, void* d_ws, size_t ws_size,
                              hipStream_t stream) {
  const float* x    = (const float*)d_in[0];
  const float* plw  = (const float*)d_in[1];
  const float* plb  = (const float*)d_in[2];
  const float* embW = (const float*)d_in[3];
  const float* embB = (const float*)d_in[4];
  const float* gW   = (const float*)d_in[5];
  const float* gB   = (const float*)d_in[6];
  const float* eW1  = (const float*)d_in[7];
  const float* eB1  = (const float*)d_in[8];
  const float* eW2  = (const float*)d_in[9];
  const float* eB2  = (const float*)d_in[10];
  const float* tw1  = (const float*)d_in[11];
  const float* tb1  = (const float*)d_in[12];
  const float* tw2  = (const float*)d_in[13];
  const float* tb2  = (const float*)d_in[14];
  // d_in[15] = top_k (==2, semantics hardcoded in gate/scatter)
  float* out = (float*)d_out;

  char* ws = (char*)d_ws;
  _Float16* flat16 = (_Float16*)(ws + OFF_FLAT16);
  _Float16* w1t    = (_Float16*)(ws + OFF_W1T);
  _Float16* w2t    = (_Float16*)(ws + OFF_W2T);
  _Float16* tw1t   = (_Float16*)(ws + OFF_TW1T);
  _Float16* h16    = (_Float16*)(ws + OFF_H16);
  float*    moe    = (float*)   (ws + OFF_MOE);
  _Float16* moe16  = (_Float16*)(ws + OFF_MOE16);
  _Float16* th16   = (_Float16*)(ws + OFF_TH16);
  float4*   route  = (float4*)  (ws + OFF_ROUTE);
  int*      rowsA  = (int*)     (ws + OFF_ROWS);
  float*    gwA    = (float*)   (ws + OFF_GW);
  int*      counts = (int*)     (ws + OFF_CNT);
  int*      offs   = (int*)     (ws + OFF_OFFS);
  int*      cursor = (int*)     (ws + OFF_CUR);

  init_kernel<<<dim3((B_ * O_ + 255) / 256), 256, 0, stream>>>(moe, counts);

  transpose_cvt_kernel<<<dim3(H_ / 64, F_ / 64, E_), 256, 0, stream>>>(eW1, w1t, F_, H_);
  transpose_cvt_kernel<<<dim3(O_ / 64, H_ / 64, E_), 256, 0, stream>>>(eW2, w2t, H_, O_);
  transpose_cvt_kernel<<<dim3(TH_ / 64, O_ / 64, T_), 256, 0, stream>>>(tw1, tw1t, O_, TH_);

  ple_embed_kernel<<<dim3(B_), 256, 0, stream>>>(x, plw, plb, embW, embB, flat16);
  gate_kernel<<<dim3(B_ / 8), 256, 0, stream>>>(flat16, gW, gB, route, counts);
  offsets_kernel<<<dim3(1), 32, 0, stream>>>(counts, offs, cursor);
  scatter_kernel<<<dim3(B_ / 256), 256, 0, stream>>>(route, cursor, rowsA, gwA);

  fc1_kernel<<<dim3(H_ / BN, B_ / BM, E_), 256, 0, stream>>>(flat16, w1t, eB1, counts, offs,
                                                             rowsA, h16);
  fc2_kernel<<<dim3(O_ / BN, B_ / BM, E_), 256, 0, stream>>>(h16, w2t, eB2, counts, offs,
                                                             rowsA, gwA, moe);

  cvt_moe_kernel<<<dim3((B_ * O_ + 255) / 256), 256, 0, stream>>>(moe, moe16);
  tower_kernel<<<dim3(TH_ / BN, B_ / BM, T_), 256, 0, stream>>>(moe16, tw1t, tb1, th16);
  final_kernel<<<dim3(B_ * T_ / 8), 256, 0, stream>>>(th16, tw2, tb2, out);
}
